// CausalSelfAttention_60997125538142
// MI455X (gfx1250) — compile-verified, matrix operand order checked by reading, not run on hardware
//
#include <hip/hip_runtime.h>

// ---------------------------------------------------------------------------
// MI455X (gfx1250) implementation: bf16 WMMA GEMMs + WMMA chunked attention.
// Wave32 everywhere; reductions use 32-lane shuffles.
// ---------------------------------------------------------------------------

#define DEVINL __device__ __forceinline__

typedef __bf16 bf16_t;
typedef __attribute__((ext_vector_type(8)))  __bf16 v8bf;
typedef __attribute__((ext_vector_type(16))) __bf16 v16bf;
typedef __attribute__((ext_vector_type(8)))  float  v8f;

constexpr int Bc = 2, Tc = 2048, Cc = 2048, Hc = 16, Nc = 128;
constexpr int Lc = 256, SINKc = 1, RPc = 64, NBc = 2, nCc = Tc / Lc, Sc = 7;
constexpr int Mc = Bc * Tc;
constexpr float NEGV = -1e30f;

DEVINL int imin(int a, int b) { return a < b ? a : b; }
DEVINL int imax(int a, int b) { return a > b ? a : b; }

DEVINL float wave_sum(float v) {
#pragma unroll
  for (int m = 16; m >= 1; m >>= 1) v += __shfl_xor(v, m, 32);
  return v;
}
DEVINL float wave_max(float v) {
#pragma unroll
  for (int m = 16; m >= 1; m >>= 1) v = fmaxf(v, __shfl_xor(v, m, 32));
  return v;
}

DEVINL v8f v8f_zero() {
  v8f z;
  for (int i = 0; i < 8; ++i) z[i] = 0.f;
  return z;
}

// Build a 16-element bf16 fragment from two contiguous 8-element (16B) runs.
DEVINL v16bf make_tile(const bf16_t* p0, const bf16_t* p1) {
  v8bf a = *(const v8bf*)p0;
  v8bf b = *(const v8bf*)p1;
  v16bf o;
#pragma unroll
  for (int j = 0; j < 8; ++j) { o[j] = a[j]; o[j + 8] = b[j]; }
  return o;
}

// Row-major [row, k] fragment loader matching the CDNA5 16-bit A/B layout:
// lane&15 selects the row within the 16-row tile (caller folds it into `row`),
// lane>>4 (hi) selects the K halves: {kb+hi*8 .. +7} and {kb+16+hi*8 .. +7}.
DEVINL v16bf load_tile(const bf16_t* base, long stride, int row, int kb, int hi) {
  const bf16_t* p = base + (long)row * stride + kb + hi * 8;
  return make_tile(p, p + 16);
}

DEVINL v8f wmma_bf(v16bf a, v16bf b, v8f c) {
  return __builtin_amdgcn_wmma_f32_16x16x32_bf16(false, a, false, b, (short)0, c,
                                                 false, false);
}

// LayerNorm over N=128 spread 4-per-lane across a wave32.
DEVINL void ln4(float* v, const float* w, const float* bias, int nbase) {
  float s = v[0] + v[1] + v[2] + v[3];
  s = wave_sum(s);
  float mean = s * (1.f / 128.f);
  float d0 = v[0] - mean, d1 = v[1] - mean, d2 = v[2] - mean, d3 = v[3] - mean;
  float var = wave_sum(d0 * d0 + d1 * d1 + d2 * d2 + d3 * d3) * (1.f / 128.f);
  float inv = rsqrtf(var + 1e-5f);
  v[0] = d0 * inv * w[nbase + 0] + bias[nbase + 0];
  v[1] = d1 * inv * w[nbase + 1] + bias[nbase + 1];
  v[2] = d2 * inv * w[nbase + 2] + bias[nbase + 2];
  v[3] = d3 * inv * w[nbase + 3] + bias[nbase + 3];
}

DEVINL void rope4(float* v, const float* cosb, const float* sinb, int t, int nbase) {
#pragma unroll
  for (int p = 0; p < 2; ++p) {
    int n0 = nbase + 2 * p;
    float a = v[2 * p], b = v[2 * p + 1];
    v[2 * p]     = cosb[t * Nc + n0] * a + sinb[t * Nc + n0] * b;
    v[2 * p + 1] = cosb[t * Nc + n0 + 1] * b + sinb[t * Nc + n0 + 1] * a;
  }
}

// ---------------------------------------------------------------------------
// fp32 -> bf16 conversion
// ---------------------------------------------------------------------------
__global__ void f2bf_kernel(const float* __restrict__ in, bf16_t* __restrict__ out,
                            long n) {
  long i = (long)blockIdx.x * blockDim.x + threadIdx.x;
  if (i < n) out[i] = (bf16_t)in[i];
}

// ---------------------------------------------------------------------------
// C[M,Nn] = A[M,K] @ W[Nn,K]^T (+ add), bf16 WMMA, f32 accumulate.
// Block = 128 threads (4 waves); each wave owns a 16x64 output strip.
// ---------------------------------------------------------------------------
__global__ void gemm_bf16_xWt(const bf16_t* __restrict__ A, const bf16_t* __restrict__ W,
                              const float* __restrict__ add, float* __restrict__ Cout,
                              int M, int Nn, int K) {
  int lane = threadIdx.x & 31, wave = threadIdx.x >> 5;
  int r = lane & 15, hi = (lane >> 4) & 1;
  int m_base = blockIdx.x * 64 + wave * 16;
  int n_base = blockIdx.y * 64;
  v8f acc[4];
#pragma unroll
  for (int j = 0; j < 4; ++j) acc[j] = v8f_zero();
  for (int kb = 0; kb < K; kb += 32) {
    __builtin_prefetch(A + (long)(m_base + r) * K + kb + 256, 0, 0);
    v16bf a = load_tile(A, K, m_base + r, kb, hi);
#pragma unroll
    for (int j = 0; j < 4; ++j) {
      v16bf bt = load_tile(W, K, n_base + j * 16 + r, kb, hi);
      acc[j] = wmma_bf(a, bt, acc[j]);
    }
  }
#pragma unroll
  for (int j = 0; j < 4; ++j) {
#pragma unroll
    for (int rr = 0; rr < 8; ++rr) {
      int m = m_base + rr + hi * 8;
      int nn = n_base + j * 16 + r;
      float v = acc[j][rr];
      if (add) v += add[(long)m * Nn + nn];
      Cout[(long)m * Nn + nn] = v;
    }
  }
}

// ---------------------------------------------------------------------------
// LoRA: t1 = x @ lora_v1 ; t2 = t1 @ lora_v2  (small, VALU fp32)
// ---------------------------------------------------------------------------
__global__ void lora1_kernel(const float* __restrict__ x, const float* __restrict__ lv1,
                             float* __restrict__ t1) {
  long i = (long)blockIdx.x * blockDim.x + threadIdx.x;
  if (i >= (long)Mc * 64) return;
  int j = (int)(i & 63);
  long m = i >> 6;
  const float* xr = x + m * Cc;
  float acc = 0.f;
  for (int k = 0; k < Cc; ++k) acc += xr[k] * lv1[(long)k * 64 + j];
  t1[i] = acc;
}
__global__ void lora2_kernel(const float* __restrict__ t1, const float* __restrict__ lv2,
                             float* __restrict__ t2) {
  long i = (long)blockIdx.x * blockDim.x + threadIdx.x;
  if (i >= (long)Mc * Cc) return;
  int c = (int)(i % Cc);
  long m = i / Cc;
  const float* tr = t1 + m * 64;
  float acc = 0.f;
#pragma unroll
  for (int j = 0; j < 64; ++j) acc += tr[j] * lv2[(long)j * Cc + c];
  t2[i] = acc;
}

// ---------------------------------------------------------------------------
// kw = 1+elu(x @ W_kw^T), lsd = -softplus(x @ W_sd^T); one wave per (b,t,h).
// ---------------------------------------------------------------------------
__global__ void kwlsd_kernel(const float* __restrict__ x, const float* __restrict__ Wkw,
                             const float* __restrict__ Wsd, float* __restrict__ kwB,
                             float* __restrict__ lsdB) {
  int lane = threadIdx.x & 31;
  int gw = blockIdx.x * (blockDim.x >> 5) + (threadIdx.x >> 5);
  int h = gw % Hc;
  int t = (gw / Hc) % Tc;
  int b = gw / (Hc * Tc);
  const float* xr = x + (long)(b * Tc + t) * Cc;
  float a1 = 0.f, a2 = 0.f;
  for (int i = lane; i < Cc; i += 32) {
    float xv = xr[i];
    a1 += xv * Wkw[(long)h * Cc + i];
    a2 += xv * Wsd[(long)h * Cc + i];
  }
  a1 = wave_sum(a1);
  a2 = wave_sum(a2);
  if (lane == 0) {
    float e = (a1 > 0.f) ? a1 : (__expf(a1) - 1.f);
    long o = (long)(b * Hc + h) * Tc + t;
    kwB[o] = 1.f + e;
    float sp = (a2 > 30.f) ? a2 : log1pf(__expf(a2));
    lsdB[o] = -sp;
  }
}

// ---------------------------------------------------------------------------
// Fused: v-gate, token-shift, LN, rope; emits q/k bf16 [B,H,T,N], k/v fp32,
// and transposed bf16 v [B,H,N,T]. One wave per (b,t,h).
// ---------------------------------------------------------------------------
__global__ void fuse_kernel(const float* __restrict__ qlin, const float* __restrict__ klin,
                            const float* __restrict__ vlin, const float* __restrict__ t2,
                            const float* __restrict__ v0, const float* __restrict__ v1_in,
                            const float* __restrict__ smask, const float* __restrict__ xq,
                            const float* __restrict__ xk, const float* __restrict__ xv,
                            const float* __restrict__ lnqw, const float* __restrict__ lnqb,
                            const float* __restrict__ lnkw, const float* __restrict__ lnkb,
                            const float* __restrict__ cosb, const float* __restrict__ sinb,
                            bf16_t* __restrict__ q_bf, bf16_t* __restrict__ k_bf,
                            float* __restrict__ kf, float* __restrict__ vf,
                            bf16_t* __restrict__ vT) {
  int lane = threadIdx.x & 31;
  int gw = blockIdx.x * (blockDim.x >> 5) + (threadIdx.x >> 5);
  int h = gw % Hc;
  int t = (gw / Hc) % Tc;
  int b = gw / (Hc * Tc);
  long row = (long)(b * Tc + t) * Cc;
  long prow = row - Cc;
  float sm = smask[t];
  bool hasPrev = (t > 0);
  int nb = lane * 4;
  int cb = h * Nc + nb;
  float q[4], k[4], v[4];
#pragma unroll
  for (int j = 0; j < 4; ++j) {
    int cc = cb + j;
    float qc = qlin[row + cc];
    float kc = klin[row + cc];
    float g = 1.f / (1.f + __expf(-(v0[cc] + t2[row + cc])));
    float vl = vlin[row + cc];
    float vc = vl + (v1_in[row + cc] - vl) * g;
    float qp = 0.f, kp = 0.f, vp = 0.f;
    if (hasPrev) {
      qp = qlin[prow + cc];
      kp = klin[prow + cc];
      float gp = 1.f / (1.f + __expf(-(v0[cc] + t2[prow + cc])));
      float vpl = vlin[prow + cc];
      vp = vpl + (v1_in[prow + cc] - vpl) * gp;
    }
    q[j] = qc + xq[cc] * (qp - qc) * sm;
    k[j] = kc + xk[cc] * (kp - kc) * sm;
    v[j] = vc + xv[cc] * (vp - vc) * sm;
  }
  ln4(q, lnqw, lnqb, nb);
  ln4(k, lnkw, lnkb, nb);
  rope4(q, cosb, sinb, t, nb);
  rope4(k, cosb, sinb, t, nb);
  long ho = ((long)(b * Hc + h) * Tc + t) * Nc + nb;
#pragma unroll
  for (int j = 0; j < 4; ++j) {
    q_bf[ho + j] = (bf16_t)q[j];
    k_bf[ho + j] = (bf16_t)k[j];
    kf[ho + j] = k[j];
    vf[ho + j] = v[j];
    vT[((long)(b * Hc + h) * Nc + nb + j) * Tc + t] = (bf16_t)v[j];
  }
}

// ---------------------------------------------------------------------------
// Memory-state kernels (deterministic; one wave per (b,h,slot)).
// ---------------------------------------------------------------------------
__global__ void mem_init(const float* __restrict__ kf, const float* __restrict__ vf,
                         const float* __restrict__ lndw, const float* __restrict__ lndb,
                         const float* __restrict__ stt, float* __restrict__ dk,
                         float* __restrict__ dv, float* __restrict__ dvlen,
                         bf16_t* __restrict__ memk, bf16_t* __restrict__ memvT) {
  int lane = threadIdx.x & 31;
  int gw = blockIdx.x * (blockDim.x >> 5) + (threadIdx.x >> 5);
  int l = gw % Lc;
  int h = (gw / Lc) % Hc;
  int b = gw / (Lc * Hc);
  long bh = (long)b * Hc + h;
  int nb = lane * 4;
  long src = (bh * Tc + l) * Nc + nb;  // chunk 0, t = l
  float z[4], dvv[4];
#pragma unroll
  for (int j = 0; j < 4; ++j) {
    int n = nb + j;
    z[j] = (n < RPc) ? 0.f : kf[src + j];
    dvv[j] = vf[src + j];
  }
  ln4(z, lndw, lndb, nb);
  float nr = wave_sum(dvv[0] * dvv[0] + dvv[1] * dvv[1] + dvv[2] * dvv[2] + dvv[3] * dvv[3]);
  float dlen = sqrtf(nr);
  long slot = (bh * Lc + l) * Nc + nb;
#pragma unroll
  for (int j = 0; j < 4; ++j) { dk[slot + j] = z[j]; dv[slot + j] = dvv[j]; }
  if (lane == 0) dvlen[bh * Lc + l] = dlen;
  float st = stt[h];
  float nrm = fmaxf(dlen, 1e-12f);
  long mrow = (bh * 8 + 0) * (long)Lc * Nc + (long)l * Nc + nb;
#pragma unroll
  for (int j = 0; j < 4; ++j) {
    memk[mrow + j] = (bf16_t)(z[j] * st);
    memvT[((bh * 8 + 0) * (long)Nc + nb + j) * Lc + l] = (bf16_t)(dvv[j] / nrm * dlen);
  }
}

__global__ void mem_prep(const float* __restrict__ kf, const float* __restrict__ dk,
                         const float* __restrict__ lndw, const float* __restrict__ lndb,
                         float* __restrict__ ck0, float* __restrict__ dkn, int idx) {
  int lane = threadIdx.x & 31;
  int gw = blockIdx.x * (blockDim.x >> 5) + (threadIdx.x >> 5);
  int s = gw % Lc;
  int h = (gw / Lc) % Hc;
  int b = gw / (Lc * Hc);
  long bh = (long)b * Hc + h;
  int nb = lane * 4;
  long src = (bh * Tc + (long)idx * Lc + s) * Nc + nb;
  float z[4];
#pragma unroll
  for (int j = 0; j < 4; ++j) {
    int n = nb + j;
    z[j] = (n < RPc) ? 0.f : kf[src + j];
  }
  ln4(z, lndw, lndb, nb);
  long slot = (bh * Lc + s) * Nc + nb;
#pragma unroll
  for (int j = 0; j < 4; ++j) ck0[slot + j] = z[j];
  float d[4];
#pragma unroll
  for (int j = 0; j < 4; ++j) d[j] = dk[slot + j];
  ln4(d, lndw, lndb, nb);
#pragma unroll
  for (int j = 0; j < 4; ++j) dkn[slot + j] = d[j];
}

__global__ void mem_argmax(const float* __restrict__ ck0, const float* __restrict__ dkn,
                           int* __restrict__ am) {
  int lane = threadIdx.x & 31;
  int gw = blockIdx.x * (blockDim.x >> 5) + (threadIdx.x >> 5);
  int l = gw % Lc;
  int h = (gw / Lc) % Hc;
  int b = gw / (Lc * Hc);
  long bh = (long)b * Hc + h;
  int nb = lane * 4;
  long cs = (bh * Lc + l) * Nc + nb;
  float c0 = ck0[cs], c1 = ck0[cs + 1], c2 = ck0[cs + 2], c3 = ck0[cs + 3];
  float best = -3.4e38f;
  int bi = 0;
  for (int m = 0; m < Lc; ++m) {
    const float* dr = dkn + (bh * Lc + m) * Nc + nb;
    float p = c0 * dr[0] + c1 * dr[1] + c2 * dr[2] + c3 * dr[3];
    p = wave_sum(p);
    if (m < SINKc) p = NEGV;
    if (p > best) { best = p; bi = m; }
  }
  if (lane == 0) am[bh * Lc + l] = bi;
}

__global__ void mem_update(const float* __restrict__ ck0, const float* __restrict__ vf,
                           const float* __restrict__ kwB, const float* __restrict__ lsdB,
                           const int* __restrict__ am, const float* __restrict__ stt,
                           const float* __restrict__ dvlen, float* __restrict__ dk,
                           float* __restrict__ dv, bf16_t* __restrict__ memk,
                           bf16_t* __restrict__ memvT, int idx, int snap) {
  int lane = threadIdx.x & 31;
  int gw = blockIdx.x * (blockDim.x >> 5) + (threadIdx.x >> 5);
  int m = gw % Lc;
  int h = (gw / Lc) % Hc;
  int b = gw / (Lc * Hc);
  long bh = (long)b * Hc + h;
  int nb = lane * 4;
  float ls = 0.f, ak[4] = {0.f, 0.f, 0.f, 0.f}, av[4] = {0.f, 0.f, 0.f, 0.f};
  for (int l = 0; l < Lc; ++l) {
    if (am[bh * Lc + l] == m) {
      long t = (long)idx * Lc + l;
      float w = kwB[bh * Tc + t];
      ls += lsdB[bh * Tc + t];
      const float* cr = ck0 + (bh * Lc + l) * Nc + nb;
      const float* vr = vf + (bh * Tc + t) * Nc + nb;
#pragma unroll
      for (int j = 0; j < 4; ++j) {
        ak[j] += cr[j] * w;
        av[j] += vr[j] * w;
      }
    }
  }
  long slot = (bh * Lc + m) * Nc + nb;
  float e = __expf(ls);
  float nk[4], nv[4];
#pragma unroll
  for (int j = 0; j < 4; ++j) {
    nk[j] = dk[slot + j] + ak[j];
    nv[j] = dv[slot + j] * e + av[j];
    dk[slot + j] = nk[j];
    dv[slot + j] = nv[j];
  }
  float nr = wave_sum(nv[0] * nv[0] + nv[1] * nv[1] + nv[2] * nv[2] + nv[3] * nv[3]);
  float nrm = fmaxf(sqrtf(nr), 1e-12f);
  float dlen = dvlen[bh * Lc + m];
  float st = stt[h];
  long mrow = (bh * 8 + snap) * (long)Lc * Nc + (long)m * Nc + nb;
#pragma unroll
  for (int j = 0; j < 4; ++j) {
    memk[mrow + j] = (bf16_t)(nk[j] * st);
    memvT[((bh * 8 + snap) * (long)Nc + nb + j) * Lc + m] = (bf16_t)(nv[j] / nrm * dlen);
  }
}

// ---------------------------------------------------------------------------
// Chunked attention. One block (8 waves) per (b,h,chunk). For each 16-row q
// strip: WMMA scores (48 col tiles) -> LDS bf16 -> softmax -> WMMA p@V.
// Cols: [0,512) local (t = (c-1)*L + col), [512,768) memory slots; sink col
// handled scalar. LDS total ~50KB.
// ---------------------------------------------------------------------------
__global__ void attn_kernel(const bf16_t* __restrict__ q_bf, const bf16_t* __restrict__ k_bf,
                            const bf16_t* __restrict__ vT, const float* __restrict__ vf,
                            const bf16_t* __restrict__ memk, const bf16_t* __restrict__ memvT,
                            const float* __restrict__ front, float* __restrict__ y) {
  __shared__ __attribute__((aligned(16))) bf16_t sS[16 * 784];
  __shared__ __attribute__((aligned(16))) bf16_t sP[16 * 784];
  __shared__ float s_rs[16], s_ps[16];
  int lane = threadIdx.x & 31, wave = threadIdx.x >> 5;
  int r = lane & 15, hi = (lane >> 4) & 1;
  int c = blockIdx.x % nCc;
  int h = (blockIdx.x / nCc) % Hc;
  int b = blockIdx.x / (nCc * Hc);
  long bh = (long)b * Hc + h;
  const bf16_t* qb = q_bf + bh * (long)Tc * Nc;
  const bf16_t* kbp = k_bf + bh * (long)Tc * Nc;
  int miC = imin(imax(c - NBc, 0), Sc);
  const bf16_t* mkb = memk + (bh * 8 + miC) * (long)Lc * Nc;
  const bf16_t* vtb = vT + bh * (long)Nc * Tc;
  const bf16_t* mvtb = memvT + (bh * 8 + miC) * (long)Nc * Lc;
  const float scale = 0.08838834764831845f;  // 1/sqrt(128)
  float ft = front[h];
  float scl = scale * ft, scm = scale;

  for (int strip = 0; strip < 16; ++strip) {
    int q0 = c * Lc + strip * 16;
    v16bf aq[4];
#pragma unroll
    for (int kk = 0; kk < 4; ++kk) aq[kk] = load_tile(qb, Nc, q0 + r, kk * 32, hi);

    // ---- stage 1: score tiles (48 of 16x16), masked, -> LDS bf16
    for (int jj = 0; jj < 6; ++jj) {
      int j = wave + jj * 8;  // 0..47
      v8f acc = v8f_zero();
      int colbase = j * 16;
      bool ismem = (j >= 32);
      const bf16_t* bb = ismem ? mkb : kbp;
      int brow = ismem ? (colbase - 512 + r) : ((c - 1) * Lc + colbase + r);
      if (brow < 0) brow = 0;  // masked region for c==0; value irrelevant
#pragma unroll
      for (int kk = 0; kk < 4; ++kk) {
        v16bf bt = load_tile(bb, Nc, brow, kk * 32, hi);
        acc = wmma_bf(aq[kk], bt, acc);
      }
      float mult = ismem ? scm : scl;
      int col = colbase + r;
#pragma unroll
      for (int rr = 0; rr < 8; ++rr) {
        int m = rr + hi * 8;
        int lloc = strip * 16 + m;
        float s = acc[rr] * mult;
        bool valid = ismem ? (c >= NBc)
                           : ((col < Lc) ? (c > 0) : ((col - Lc) <= lloc));
        if (!valid) s = NEGV;
        sS[m * 784 + col] = (bf16_t)s;
      }
    }
    __syncthreads();

    // ---- stage 2: softmax per row (2 rows per wave) + sink column
#pragma unroll
    for (int rr2 = 0; rr2 < 2; ++rr2) {
      int m = wave * 2 + rr2;
      int tq = q0 + m;
      const bf16_t* qrow = qb + (long)tq * Nc;
      float sd = 0.f;
#pragma unroll
      for (int j = 0; j < 4; ++j) {
        int n = lane * 4 + j;
        sd += (float)qrow[n] * (float)kbp[n];  // k at t=0 (sink)
      }
      sd = wave_sum(sd) * scl;
      if (c < NBc) sd = NEGV;
      float mx = sd;
      for (int col = lane; col < 768; col += 32) mx = fmaxf(mx, (float)sS[m * 784 + col]);
      mx = wave_max(mx);
      float rs = 0.f;
      for (int col = lane; col < 768; col += 32) {
        float p = __expf((float)sS[m * 784 + col] - mx);
        sP[m * 784 + col] = (bf16_t)p;
        rs += p;
      }
      rs = wave_sum(rs);
      float ps = __expf(sd - mx);
      if (lane == 0) {
        s_rs[m] = rs + ps;
        s_ps[m] = ps;
      }
    }
    __syncthreads();

    // ---- stage 3: O = P @ V (one 16x16 head-dim tile per wave, k over 768)
    {
      int jd = wave;  // 0..7 head-dim tiles
      v8f acc = v8f_zero();
      int drow = jd * 16 + r;  // head dim for B lanes
      for (int kb2 = 0; kb2 < 768; kb2 += 32) {
        v16bf at = load_tile((const bf16_t*)sP, 784, r, kb2, hi);  // LDS reads
        v16bf bt;
        if (kb2 < 512) {
          int t0 = (c - 1) * Lc + kb2 + hi * 8;
          int t1 = t0 + 16;
          if (t0 < 0) t0 = 0;
          if (t1 < 0) t1 = 0;
          const bf16_t* base = vtb + (long)drow * Tc;
          bt = make_tile(base + t0, base + t1);
        } else {
          const bf16_t* base = mvtb + (long)drow * Lc + (kb2 - 512) + hi * 8;
          bt = make_tile(base, base + 16);
        }
        acc = wmma_bf(at, bt, acc);
      }
      float vs = vf[bh * (long)Tc * Nc + drow];  // v sink (t=0)
#pragma unroll
      for (int rr = 0; rr < 8; ++rr) {
        int m = rr + hi * 8;
        float val = (acc[rr] + s_ps[m] * vs) / s_rs[m];
        int tq = q0 + m;
        y[((long)b * Tc + tq) * Cc + h * Nc + drow] = val;
      }
    }
    __syncthreads();
  }
}

// ---------------------------------------------------------------------------
// Host orchestration
// ---------------------------------------------------------------------------
extern "C" void kernel_launch(void* const* d_in, const int* in_sizes, int n_in,
                              void* d_out, int out_size, void* d_ws, size_t ws_size,
                              hipStream_t stream) {
  const float* residual = (const float*)d_in[0];
  const float* x = (const float*)d_in[1];
  const float* v1_in = (const float*)d_in[2];
  const float* smask = (const float*)d_in[3];
  const float* cosb = (const float*)d_in[4];
  const float* sinb = (const float*)d_in[5];
  const float* Wq = (const float*)d_in[6];
  const float* Wk = (const float*)d_in[7];
  const float* Wv = (const float*)d_in[8];
  const float* Wp = (const float*)d_in[9];
  const float* xq = (const float*)d_in[10];
  const float* xk = (const float*)d_in[11];
  const float* xv = (const float*)d_in[12];
  const float* lnqw = (const float*)d_in[13];
  const float* lnqb = (const float*)d_in[14];
  const float* lnkw = (const float*)d_in[15];
  const float* lnkb = (const float*)d_in[16];
  const float* lndw = (const float*)d_in[17];
  const float* lndb = (const float*)d_in[18];
  const float* v0 = (const float*)d_in[19];
  const float* lora_v1 = (const float*)d_in[20];
  const float* lora_v2 = (const float*)d_in[21];
  const float* Wkw = (const float*)d_in[22];
  const float* Wsd = (const float*)d_in[23];
  const float* front = (const float*)d_in[24];
  const float* stt = (const float*)d_in[25];
  // d_in[26] = chunk_indices (arange(1, 8)); mirrored on host below.

  char* wsp = (char*)d_ws;
  size_t off = 0;
  auto alloc = [&](size_t bytes) -> void* {
    void* p = wsp + off;
    off = (off + bytes + 255) & ~(size_t)255;
    return p;
  };
  bf16_t* x_bf = (bf16_t*)alloc((size_t)Mc * Cc * 2);
  bf16_t* wq_bf = (bf16_t*)alloc((size_t)Cc * Cc * 2);
  bf16_t* wk_bf = (bf16_t*)alloc((size_t)Cc * Cc * 2);
  bf16_t* wv_bf = (bf16_t*)alloc((size_t)Cc * Cc * 2);
  bf16_t* wp_bf = (bf16_t*)alloc((size_t)Cc * Cc * 2);
  float* qlin = (float*)alloc((size_t)Mc * Cc * 4);
  float* klin = (float*)alloc((size_t)Mc * Cc * 4);
  float* vlin = (float*)alloc((size_t)Mc * Cc * 4);
  float* t1 = (float*)alloc((size_t)Mc * 64 * 4);
  float* t2 = (float*)alloc((size_t)Mc * Cc * 4);
  float* kwB = (float*)alloc((size_t)Bc * Hc * Tc * 4);
  float* lsdB = (float*)alloc((size_t)Bc * Hc * Tc * 4);
  bf16_t* q_bf = (bf16_t*)alloc((size_t)Mc * Cc * 2);
  bf16_t* k_bf = (bf16_t*)alloc((size_t)Mc * Cc * 2);
  float* kf = (float*)alloc((size_t)Mc * Cc * 4);
  float* vfp = (float*)alloc((size_t)Mc * Cc * 4);
  bf16_t* vT = (bf16_t*)alloc((size_t)Mc * Cc * 2);
  float* dk = (float*)alloc((size_t)Bc * Hc * Lc * Nc * 4);
  float* dv = (float*)alloc((size_t)Bc * Hc * Lc * Nc * 4);
  float* dkn = (float*)alloc((size_t)Bc * Hc * Lc * Nc * 4);
  float* ck0 = (float*)alloc((size_t)Bc * Hc * Lc * Nc * 4);
  float* dvlen = (float*)alloc((size_t)Bc * Hc * Lc * 4);
  int* am = (int*)alloc((size_t)Bc * Hc * Lc * 4);
  bf16_t* memk = (bf16_t*)alloc((size_t)Bc * Hc * 8 * Lc * Nc * 2);
  bf16_t* memvT = (bf16_t*)alloc((size_t)Bc * Hc * 8 * Lc * Nc * 2);
  float* y = qlin;      // qlin dead after fuse
  bf16_t* y_bf = x_bf;  // x_bf dead after the three projection GEMMs

  const int TPB = 256;
  auto cdiv = [](long a, long b) { return (int)((a + b - 1) / b); };

  f2bf_kernel<<<cdiv((long)Mc * Cc, TPB), TPB, 0, stream>>>(x, x_bf, (long)Mc * Cc);
  f2bf_kernel<<<cdiv((long)Cc * Cc, TPB), TPB, 0, stream>>>(Wq, wq_bf, (long)Cc * Cc);
  f2bf_kernel<<<cdiv((long)Cc * Cc, TPB), TPB, 0, stream>>>(Wk, wk_bf, (long)Cc * Cc);
  f2bf_kernel<<<cdiv((long)Cc * Cc, TPB), TPB, 0, stream>>>(Wv, wv_bf, (long)Cc * Cc);
  f2bf_kernel<<<cdiv((long)Cc * Cc, TPB), TPB, 0, stream>>>(Wp, wp_bf, (long)Cc * Cc);

  dim3 gg(Mc / 64, Cc / 64);
  gemm_bf16_xWt<<<gg, 128, 0, stream>>>(x_bf, wq_bf, nullptr, qlin, Mc, Cc, Cc);
  gemm_bf16_xWt<<<gg, 128, 0, stream>>>(x_bf, wk_bf, nullptr, klin, Mc, Cc, Cc);
  gemm_bf16_xWt<<<gg, 128, 0, stream>>>(x_bf, wv_bf, nullptr, vlin, Mc, Cc, Cc);

  lora1_kernel<<<cdiv((long)Mc * 64, TPB), TPB, 0, stream>>>(x, lora_v1, t1);
  lora2_kernel<<<cdiv((long)Mc * Cc, TPB), TPB, 0, stream>>>(t1, lora_v2, t2);

  kwlsd_kernel<<<Bc * Tc * Hc / 8, TPB, 0, stream>>>(x, Wkw, Wsd, kwB, lsdB);

  fuse_kernel<<<Bc * Tc * Hc / 8, TPB, 0, stream>>>(
      qlin, klin, vlin, t2, v0, v1_in, smask, xq, xk, xv, lnqw, lnqb, lnkw, lnkb,
      cosb, sinb, q_bf, k_bf, kf, vfp, vT);

  mem_init<<<Bc * Hc * Lc / 8, TPB, 0, stream>>>(kf, vfp, lndw, lndb, stt, dk, dv,
                                                 dvlen, memk, memvT);
  for (int i = 0; i < Sc; ++i) {
    int idx = i + 1;  // chunk_indices = arange(1, T/L)
    mem_prep<<<Bc * Hc * Lc / 8, TPB, 0, stream>>>(kf, dk, lndw, lndb, ck0, dkn, idx);
    mem_argmax<<<Bc * Hc * Lc / 8, TPB, 0, stream>>>(ck0, dkn, am);
    mem_update<<<Bc * Hc * Lc / 8, TPB, 0, stream>>>(ck0, vfp, kwB, lsdB, am, stt,
                                                     dvlen, dk, dv, memk, memvT, idx,
                                                     i + 1);
  }

  attn_kernel<<<Bc * Hc * nCc, TPB, 0, stream>>>(q_bf, k_bf, vT, vfp, memk, memvT,
                                                 front, y);

  f2bf_kernel<<<cdiv((long)Mc * Cc, TPB), TPB, 0, stream>>>(y, y_bf, (long)Mc * Cc);
  gemm_bf16_xWt<<<gg, 128, 0, stream>>>(y_bf, wp_bf, residual, (float*)d_out, Mc, Cc,
                                        Cc);
  (void)in_sizes; (void)n_in; (void)out_size; (void)ws_size;
}
